// RomHog_fA_P4CNN_31190052504416
// MI455X (gfx1250) — matrix operand order repaired and checked by reading, not compiled, not run on hardware
//
#include <hip/hip_runtime.h>
#include <math.h>

typedef __attribute__((ext_vector_type(16))) _Float16 v16h;
typedef __attribute__((ext_vector_type(8)))  float    v8f;

#define C_O   10
#define C_TOT 40

__device__ __forceinline__ void rot_src(int r, int ks, int u, int v, int& su, int& sv) {
  switch (r & 3) {
    case 0:  su = u;          sv = v;          break;
    case 1:  su = v;          sv = ks - 1 - u; break;   // np.rot90 CCW once: out[u][v] = in[v][k-1-u]
    case 2:  su = ks - 1 - u; sv = ks - 1 - v; break;
    default: su = ks - 1 - v; sv = u;          break;
  }
}

__device__ __forceinline__ float sigmoidf(float x) { return 1.0f / (1.0f + __expf(-x)); }

// ---------------- weight prep ----------------

// Build B-fragments for V_WMMA_F32_16X16X32_F16 directly from c{i}_w [10,10,4,ks,ks].
// Bpack linear index: nt*(chunks*512) + ch*512 + lane*16 + e
// B layout (32x16 f16): lane<16 -> N=lane, K=e ; lane>=16 -> N=lane-16, K=16+e
__global__ void pack_weights_gg(const float* __restrict__ w, _Float16* __restrict__ Bpack,
                                int ksize, int Kreal, int chunks) {
  long idx = blockIdx.x * (long)blockDim.x + threadIdx.x;
  long total = 3L * chunks * 32 * 16;
  if (idx >= total) return;
  int e    = idx & 15;
  int lane = (idx >> 4) & 31;
  int ch   = (int)((idx >> 9) % chunks);
  int nt   = (int)(idx / ((long)chunks * 512));
  int n = nt * 16 + (lane & 15);
  int k = ch * 32 + ((lane >> 4) * 16) + e;
  float val = 0.f;
  if (n < C_TOT && k < Kreal) {
    int r = n / C_O, o = n % C_O;
    int ks2 = ksize * ksize;
    int cin = k / ks2; int rem = k - cin * ks2;
    int u = rem / ksize, v = rem - (rem / ksize) * ksize;
    int i = cin >> 2, rp = cin & 3;
    int rq = (rp - r) & 3;
    int su, sv; rot_src(r, ksize, u, v, su, sv);
    val = w[(((size_t)(o * C_O + i) * 4 + rq) * ksize + su) * ksize + sv];
  }
  Bpack[idx] = (_Float16)val;
}

// awall[r][ic][u][v] (4x8x7x7) from a{i}_w [1,2,4,7,7]
__global__ void prep_awall_kernel(const float* __restrict__ aw, float* __restrict__ awall) {
  int idx = blockIdx.x * blockDim.x + threadIdx.x;
  if (idx >= 4 * 8 * 49) return;
  int v = idx % 7; int u = (idx / 7) % 7; int ic = (idx / 49) % 8; int r = idx / 392;
  int i = ic >> 2, rp = ic & 3;
  int rq = (rp - r) & 3;
  int su, sv; rot_src(r, 7, u, v, su, sv);
  awall[idx] = aw[((size_t)(i * 4 + rq) * 7 + su) * 7 + sv];
}

// ---------------- layer 1 (lifting) ----------------

__global__ void att1_gate_kernel(const float* __restrict__ x, const float* __restrict__ a1w,
                                 float* __restrict__ xg, int Btot) {
  long idx = blockIdx.x * (long)blockDim.x + threadIdx.x;
  long n = (long)Btot * 28 * 28;
  if (idx >= n) return;
  int w = idx % 28; int h = (idx / 28) % 28; int b = (int)(idx / 784);
  const float* xb = x + (size_t)b * 784;
  float acc = 0.f;
  for (int u = 0; u < 7; ++u) {
    int hh = h + u - 3; if (hh < 0 || hh >= 28) continue;
    for (int v = 0; v < 7; ++v) {
      int ww = w + v - 3; if (ww < 0 || ww >= 28) continue;
      acc += xb[hh * 28 + ww] * (a1w[u * 7 + v] + a1w[49 + u * 7 + v]);
    }
  }
  xg[idx] = xb[h * 28 + w] * sigmoidf(acc);
}

__global__ void conv1_lift_kernel(const float* __restrict__ xg, const float* __restrict__ c1w,
                                  float* __restrict__ y, int Btot) {
  long idx = blockIdx.x * (long)blockDim.x + threadIdx.x;
  long n = (long)Btot * C_TOT * 26 * 26;
  if (idx >= n) return;
  int ow = idx % 26; long t = idx / 26; int oh = t % 26; t /= 26; int co = t % C_TOT; int b = (int)(t / C_TOT);
  int r = co / C_O, o = co % C_O;
  const float* xb = xg + (size_t)b * 784;
  float acc = 0.f;
#pragma unroll
  for (int u = 0; u < 3; ++u)
#pragma unroll
    for (int v = 0; v < 3; ++v) {
      int su, sv; rot_src(r, 3, u, v, su, sv);
      acc += xb[(oh + u) * 28 + (ow + v)] * c1w[(size_t)o * 9 + su * 3 + sv];
    }
  y[idx] = acc;
}

// ---------------- attention (GG) ----------------

__global__ void pool_gg_kernel(const _Float16* __restrict__ x, float* __restrict__ pooled,
                               int Btot, int H, int W) {
  long idx = blockIdx.x * (long)blockDim.x + threadIdx.x;
  long n = (long)Btot * 4 * H * W;
  if (idx >= n) return;
  int w = idx % W; long t = idx / W; int h = t % H; t /= H; int r = t % 4; int b = (int)(t / 4);
  float s = 0.f, mx = -__builtin_huge_valf();
#pragma unroll
  for (int o = 0; o < C_O; ++o) {
    float v = (float)x[(((size_t)b * C_TOT + o * 4 + r) * H + h) * W + w];
    s += v; mx = fmaxf(mx, v);
  }
  size_t sp = (size_t)H * W;
  pooled[(((size_t)b * 2 + 0) * 4 + r) * sp + h * W + w] = s * 0.1f;
  pooled[(((size_t)b * 2 + 1) * 4 + r) * sp + h * W + w] = mx;
}

__global__ void attgate_gg_kernel(const _Float16* __restrict__ x, const float* __restrict__ pooled,
                                  const float* __restrict__ awall, _Float16* __restrict__ xg,
                                  int Btot, int H, int W) {
  long idx = blockIdx.x * (long)blockDim.x + threadIdx.x;
  long n = (long)Btot * 4 * H * W;
  if (idx >= n) return;
  int w = idx % W; long t = idx / W; int h = t % H; t /= H; int r = t % 4; int b = (int)(t / 4);
  size_t sp = (size_t)H * W;
  float acc = 0.f;
  for (int ic = 0; ic < 8; ++ic) {
    const float* pch = pooled + (((size_t)b * 2 + (ic >> 2)) * 4 + (ic & 3)) * sp;
    const float* wk  = awall + ((size_t)r * 8 + ic) * 49;
    for (int u = 0; u < 7; ++u) {
      int hh = h + u - 3; if (hh < 0 || hh >= H) continue;
      for (int v = 0; v < 7; ++v) {
        int ww = w + v - 3; if (ww < 0 || ww >= W) continue;
        acc += pch[(size_t)hh * W + ww] * wk[u * 7 + v];
      }
    }
  }
  float att = sigmoidf(acc);
#pragma unroll
  for (int o = 0; o < C_O; ++o) {
    size_t off = (((size_t)b * C_TOT + o * 4 + r) * H + h) * W + w;
    xg[off] = (_Float16)((float)x[off] * att);
  }
}

// ---------------- WMMA implicit-GEMM conv ----------------
// One wave per block computes a 32(M) x 48(N) tile: two A fragments share three B fragments,
// 6 WMMAs per 32-wide K chunk. KSIZE/KREAL/CHUNKS are compile-time so all per-element K
// decomposition is magic-multiply/shift (no runtime integer division in the hot loop).
// A (f16 16x32): lane<16 -> M=lane, K 0..15 ; lane>=16 -> M=lane-16, K 16..31.
template <int KSIZE, int KREAL, int CHUNKS>
__global__ void conv_wmma_kernel(const _Float16* __restrict__ xg, const _Float16* __restrict__ Bpack,
                                 float* __restrict__ y, int Btot, int H, int W, int OH, int OW) {
  constexpr int KS2 = KSIZE * KSIZE;
  const int lane = threadIdx.x;
  const long M = (long)Btot * OH * OW;
  const long m0 = (long)blockIdx.x * 32;
  const int kbase = (lane >= 16) ? 16 : 0;

  // decompose the two M rows this lane sources for A (once, outside the K loop)
  const _Float16* xb[2];
  int oh_[2], ow_[2];
  bool mv[2];
#pragma unroll
  for (int t = 0; t < 2; ++t) {
    long mrow = m0 + t * 16 + (lane & 15);
    mv[t] = (mrow < M);
    int b = 0, oh = 0, ow = 0;
    if (mv[t]) { ow = (int)(mrow % OW); long q = mrow / OW; oh = (int)(q % OH); b = (int)(q / OH); }
    oh_[t] = oh; ow_[t] = ow;
    xb[t] = xg + (size_t)b * C_TOT * H * W;
  }

  v8f acc[2][3] = {};
  const size_t ntstride = (size_t)CHUNKS * 32 * 16;

  for (int ch = 0; ch < CHUNKS; ++ch) {
    v16h a[2];
#pragma unroll
    for (int t = 0; t < 2; ++t) {
#pragma unroll
      for (int e = 0; e < 16; ++e) {
        int kl = kbase + e;
        int k = ch * 32 + kl;
        _Float16 val = (_Float16)0.f;
        if (mv[t] && k < KREAL) {
          int cin = k / KS2; int rem = k - cin * KS2;
          int u = rem / KSIZE; int v = rem - u * KSIZE;
          val = xb[t][((size_t)cin * H + (oh_[t] + u)) * W + (ow_[t] + v)];
        }
        a[t][e] = val;
      }
    }
    const _Float16* bp = Bpack + ((size_t)ch * 32 + lane) * 16;
    v16h b0 = *(const v16h*)(bp);
    v16h b1 = *(const v16h*)(bp + ntstride);
    v16h b2 = *(const v16h*)(bp + 2 * ntstride);
#pragma unroll
    for (int t = 0; t < 2; ++t) {
      acc[t][0] = __builtin_amdgcn_wmma_f32_16x16x32_f16(false, a[t], false, b0, (short)0, acc[t][0], false, false);
      acc[t][1] = __builtin_amdgcn_wmma_f32_16x16x32_f16(false, a[t], false, b1, (short)0, acc[t][1], false, false);
      acc[t][2] = __builtin_amdgcn_wmma_f32_16x16x32_f16(false, a[t], false, b2, (short)0, acc[t][2], false, false);
    }
  }

  // D (f32 16x16): lane<16 -> N=lane, M=q ; lane>=16 -> N=lane-16, M=q+8
  const int n  = lane & 15;
  const int mr = (lane >> 4) * 8;
  const size_t cs0 = (size_t)OH * OW;
#pragma unroll
  for (int t = 0; t < 2; ++t) {
#pragma unroll
    for (int q = 0; q < 8; ++q) {
      long m = m0 + t * 16 + mr + q;
      if (m >= M) continue;
      int oww = (int)(m % OW); long tt = m / OW; int ohh = (int)(tt % OH); int bb = (int)(tt / OH);
      float* yb = y + (((size_t)bb * C_TOT) * OH + ohh) * OW + oww;
      if (n      < C_TOT) yb[(size_t)(n)      * cs0] = acc[t][0][q];
      if (16 + n < C_TOT) yb[(size_t)(16 + n) * cs0] = acc[t][1][q];
      if (32 + n < C_TOT) yb[(size_t)(32 + n) * cs0] = acc[t][2][q];
    }
  }
}

// ---------------- BatchNorm ----------------

__global__ void zero_f32(float* p, int n) {
  int i = blockIdx.x * blockDim.x + threadIdx.x;
  if (i < n) p[i] = 0.f;
}

__global__ void bn_stats_kernel(const float* __restrict__ y, float* __restrict__ stats,
                                int Btot, int OH, int OW) {
  __shared__ float sh_s[256];
  __shared__ float sh_q[256];
  const int co = blockIdx.y;
  const int o = co % C_O;
  const int sp = OH * OW;
  const long n = (long)Btot * sp;
  float s = 0.f, q = 0.f;
  for (long i = blockIdx.x * (long)blockDim.x + threadIdx.x; i < n; i += (long)gridDim.x * blockDim.x) {
    long b = i / sp; int rr = (int)(i % sp);
    float v = y[((size_t)b * C_TOT + co) * sp + rr];
    s += v; q += v * v;
  }
  sh_s[threadIdx.x] = s; sh_q[threadIdx.x] = q;
  __syncthreads();
  for (int d = 128; d > 0; d >>= 1) {
    if ((int)threadIdx.x < d) { sh_s[threadIdx.x] += sh_s[threadIdx.x + d]; sh_q[threadIdx.x] += sh_q[threadIdx.x + d]; }
    __syncthreads();
  }
  if (threadIdx.x == 0) { atomicAdd(&stats[o], sh_s[0]); atomicAdd(&stats[C_O + o], sh_q[0]); }
}

__global__ void bn_relu_kernel(const float* __restrict__ y, const float* __restrict__ stats,
                               const float* __restrict__ g, const float* __restrict__ bta,
                               _Float16* __restrict__ out, int Btot, int OH, int OW) {
  long idx = blockIdx.x * (long)blockDim.x + threadIdx.x;
  long n = (long)Btot * C_TOT * OH * OW;
  if (idx >= n) return;
  int ow = idx % OW; long t = idx / OW; int oh = t % OH; t /= OH; int co = t % C_TOT; int b = (int)(t / C_TOT);
  int o = co % C_O, r = co / C_O;
  float cnt = (float)Btot * 4.f * OH * OW;
  float mean = stats[o] / cnt;
  float var = stats[C_O + o] / cnt - mean * mean;
  float sc = g[o] * rsqrtf(var + 2e-5f);
  float val = (y[idx] - mean) * sc + bta[o];
  val = fmaxf(val, 0.f);
  out[(((size_t)b * C_TOT + o * 4 + r) * OH + oh) * OW + ow] = (_Float16)val;
}

__global__ void maxpool_kernel(const _Float16* __restrict__ x, _Float16* __restrict__ y,
                               int Btot, int H, int W) {
  int OH = H >> 1, OW = W >> 1;
  long n = (long)Btot * C_TOT * OH * OW;
  long idx = blockIdx.x * (long)blockDim.x + threadIdx.x;
  if (idx >= n) return;
  int ow = idx % OW; long t = idx / OW; int oh = t % OH; t /= OH; int c = t % C_TOT; int b = (int)(t / C_TOT);
  const _Float16* xb = x + (((size_t)b * C_TOT + c) * H + oh * 2) * W + ow * 2;
  float m = fmaxf(fmaxf((float)xb[0], (float)xb[1]), fmaxf((float)xb[W], (float)xb[W + 1]));
  y[idx] = (_Float16)m;
}

__global__ void final_max_kernel(const float* __restrict__ y7, float* __restrict__ out, int Btot) {
  int idx = blockIdx.x * blockDim.x + threadIdx.x;
  if (idx >= Btot * C_O) return;
  int o = idx % C_O, b = idx / C_O;
  float m = -__builtin_huge_valf();
#pragma unroll
  for (int r = 0; r < 4; ++r) m = fmaxf(m, y7[(size_t)b * C_TOT + r * C_O + o]);
  out[idx] = m;
}

// ---------------- host ----------------

extern "C" void kernel_launch(void* const* d_in, const int* in_sizes, int n_in,
                              void* d_out, int out_size, void* d_ws, size_t ws_size,
                              hipStream_t stream) {
  (void)in_sizes; (void)n_in; (void)out_size; (void)ws_size;
  const int B = 2048;
  const float* x   = (const float*)d_in[0];
  const float* c1w = (const float*)d_in[1];
  const float* a1w = (const float*)d_in[2];
  const float* cw[8]; const float* aw[8];
  for (int i = 2; i <= 7; ++i) { cw[i] = (const float*)d_in[3 + (i - 2) * 2]; aw[i] = (const float*)d_in[4 + (i - 2) * 2]; }
  const float* bng[7]; const float* bnb[7];
  for (int i = 1; i <= 6; ++i) { bng[i] = (const float*)d_in[15 + (i - 1) * 2]; bnb[i] = (const float*)d_in[16 + (i - 1) * 2]; }
  float* out = (float*)d_out;

  // workspace carve (peak ~490 MB)
  char* p = (char*)d_ws;
  auto take = [&](size_t bytes) { char* r = p; p += (bytes + 255) & ~(size_t)255; return r; };
  _Float16* actA = (_Float16*)take((size_t)B * 40 * 26 * 26 * 2);
  _Float16* actB = (_Float16*)take((size_t)B * 40 * 26 * 26 * 2);
  float* ybuf    = (float*)take((size_t)B * 40 * 26 * 26 * 4);
  float* pooled  = (float*)take((size_t)B * 8 * 26 * 26 * 4);
  float* xg1     = (float*)take((size_t)B * 28 * 28 * 4);
  float* stats   = (float*)take(32 * 4);
  _Float16* Bp[8];
  for (int i = 2; i <= 6; ++i) Bp[i] = (_Float16*)take(3 * 12 * 32 * 16 * 2);
  Bp[7] = (_Float16*)take(3 * 20 * 32 * 16 * 2);
  float* awall[8];
  for (int i = 2; i <= 7; ++i) awall[i] = (float*)take(4 * 8 * 49 * 4);

  auto g1 = [](long n) { return dim3((unsigned)((n + 255) / 256)); };

  // weight prep
  for (int i = 2; i <= 7; ++i)
    prep_awall_kernel<<<g1(4 * 8 * 49), 256, 0, stream>>>(aw[i], awall[i]);
  for (int i = 2; i <= 6; ++i)
    pack_weights_gg<<<g1(3L * 12 * 512), 256, 0, stream>>>(cw[i], Bp[i], 3, 360, 12);
  pack_weights_gg<<<g1(3L * 20 * 512), 256, 0, stream>>>(cw[7], Bp[7], 4, 640, 20);

  // layer 1 (lifting, f32 direct)
  att1_gate_kernel<<<g1((long)B * 28 * 28), 256, 0, stream>>>(x, a1w, xg1, B);
  conv1_lift_kernel<<<g1((long)B * 40 * 26 * 26), 256, 0, stream>>>(xg1, c1w, ybuf, B);
  zero_f32<<<1, 32, 0, stream>>>(stats, 20);
  bn_stats_kernel<<<dim3(64, 40), 256, 0, stream>>>(ybuf, stats, B, 26, 26);
  bn_relu_kernel<<<g1((long)B * 40 * 26 * 26), 256, 0, stream>>>(ybuf, stats, bng[1], bnb[1], actA, B, 26, 26);

  // layer 2: 26 -> 24, then 2x2 pool -> 12
  pool_gg_kernel<<<g1((long)B * 4 * 26 * 26), 256, 0, stream>>>(actA, pooled, B, 26, 26);
  attgate_gg_kernel<<<g1((long)B * 4 * 26 * 26), 256, 0, stream>>>(actA, pooled, awall[2], actB, B, 26, 26);
  {
    long M = (long)B * 24 * 24;
    conv_wmma_kernel<3, 360, 12><<<dim3((unsigned)((M + 31) / 32)), 32, 0, stream>>>(actB, Bp[2], ybuf, B, 26, 26, 24, 24);
  }
  zero_f32<<<1, 32, 0, stream>>>(stats, 20);
  bn_stats_kernel<<<dim3(64, 40), 256, 0, stream>>>(ybuf, stats, B, 24, 24);
  bn_relu_kernel<<<g1((long)B * 40 * 24 * 24), 256, 0, stream>>>(ybuf, stats, bng[2], bnb[2], actA, B, 24, 24);
  maxpool_kernel<<<g1((long)B * 40 * 12 * 12), 256, 0, stream>>>(actA, actB, B, 24, 24);

  _Float16* cur = actB;
  _Float16* scr = actA;

  // layers 3..6: 12->10, 10->8, 8->6, 6->4
  int Hin = 12;
  for (int li = 3; li <= 6; ++li) {
    int H = Hin, W = Hin, OH = H - 2, OW = W - 2;
    pool_gg_kernel<<<g1((long)B * 4 * H * W), 256, 0, stream>>>(cur, pooled, B, H, W);
    attgate_gg_kernel<<<g1((long)B * 4 * H * W), 256, 0, stream>>>(cur, pooled, awall[li], scr, B, H, W);
    long M = (long)B * OH * OW;
    conv_wmma_kernel<3, 360, 12><<<dim3((unsigned)((M + 31) / 32)), 32, 0, stream>>>(scr, Bp[li], ybuf, B, H, W, OH, OW);
    zero_f32<<<1, 32, 0, stream>>>(stats, 20);
    bn_stats_kernel<<<dim3(64, 40), 256, 0, stream>>>(ybuf, stats, B, OH, OW);
    bn_relu_kernel<<<g1((long)B * 40 * OH * OW), 256, 0, stream>>>(ybuf, stats, bng[li], bnb[li], cur, B, OH, OW);
    Hin = OH;
  }

  // layer 7: 4x4 kernel -> 1x1, K = 40*16 = 640, then orientation max
  pool_gg_kernel<<<g1((long)B * 4 * 4 * 4), 256, 0, stream>>>(cur, pooled, B, 4, 4);
  attgate_gg_kernel<<<g1((long)B * 4 * 4 * 4), 256, 0, stream>>>(cur, pooled, awall[7], scr, B, 4, 4);
  conv_wmma_kernel<4, 640, 20><<<dim3((unsigned)((B + 31) / 32)), 32, 0, stream>>>(scr, Bp[7], ybuf, B, 4, 4, 1, 1);
  final_max_kernel<<<g1((long)B * 10), 256, 0, stream>>>(ybuf, out, B);
}
